// graph_reconstruction_14035953123812
// MI455X (gfx1250) — matrix operand order repaired and checked by reading, not compiled, not run on hardware
//
#include <hip/hip_runtime.h>
#include <math.h>

#define NN   30000
#define NE   960000
#define HID  34
#define HBW  40     // padded bf16 h row (halfs): 80 B, 16B-aligned
#define FOUT 17
#define NDEP 6

typedef __attribute__((ext_vector_type(16))) __bf16 v16bf;
typedef __attribute__((ext_vector_type(16))) short  v16s;
typedef __attribute__((ext_vector_type(8)))  float  v8f;

// ---------- helpers ----------
static __device__ __forceinline__ unsigned short f2bf(float f) {
  unsigned u = __float_as_uint(f);
  u += 0x7fffu + ((u >> 16) & 1u);          // round-to-nearest-even
  return (unsigned short)(u >> 16);
}
// order-preserving float<->uint mapping for atomicMin/Max
static __device__ __forceinline__ unsigned fmap(float f) {
  unsigned u = __float_as_uint(f);
  return (u & 0x80000000u) ? ~u : (u | 0x80000000u);
}
static __device__ __forceinline__ float funmap(unsigned u) {
  unsigned b = (u & 0x80000000u) ? (u ^ 0x80000000u) : ~u;
  return __uint_as_float(b);
}

// ---------- degree kernels ----------
__global__ void k_count(const int* __restrict__ dst, int* __restrict__ cnt) {
  int e = blockIdx.x * blockDim.x + threadIdx.x;
  if (e < NE) atomicAdd(&cnt[dst[e]], 1);
}
__global__ void k_logsum(const int* __restrict__ cnt, float* __restrict__ lsum) {
  int n = blockIdx.x * blockDim.x + threadIdx.x;
  if (n < NN) atomicAdd(lsum, logf((float)cnt[n] + 1.0f));
}
__global__ void k_degscale(const int* __restrict__ cnt, const float* __restrict__ lsum,
                           float* __restrict__ amp, float* __restrict__ att) {
  int n = blockIdx.x * blockDim.x + threadIdx.x;
  if (n >= NN) return;
  float avg = lsum[0] / (float)NN;
  float deg = fmaxf((float)cnt[n], 1.0f);
  float ld  = logf(deg + 1.0f);
  amp[n] = ld / avg;
  att[n] = avg / ld;
}
// init h (f32, 34-wide) and its padded bf16 mirror (40-wide, pads zeroed)
__global__ void k_inith(const float* __restrict__ x, const float* __restrict__ t,
                        float* __restrict__ h, unsigned short* __restrict__ hbf) {
  int i = blockIdx.x * blockDim.x + threadIdx.x;
  if (i >= NN * HBW) return;
  int n = i / HBW, c = i % HBW;
  float v = 0.0f;
  if (c < 5)       v = x[n * 5 + c];
  else if (c == 5) v = t[n];
  hbf[i] = f2bf(v);
  if (c < HID) h[n * HID + c] = v;
}
__global__ void k_init_minmax(unsigned* __restrict__ amin, unsigned* __restrict__ amax, int total) {
  int i = blockIdx.x * blockDim.x + threadIdx.x;
  if (i >= total) return;
  amin[i] = 0xFF800000u;  // fmap(+inf)
  amax[i] = 0x007FFFFFu;  // fmap(-inf)
}

// ---------- edge pre-MLP (WMMA bf16) + scatter aggregation ----------
// Wave32 block processes TPB tiles of 16 edges.  A-tile K layout (row = edge):
//   [ h[dst] padded to 40 | h[src] padded to 40 | 16 zeros ]  -> KP = 96 always
// Both 40-half regions are 16B-aligned, so staging is 5x uint4 copies per lane
// (lane<16 copies h[dst[r]], lane>=16 copies h[src[r]]).  Weights are staged
// bf16, permuted to this K order AND pre-packed in WMMA-B lane order so each
// fragment is one contiguous 32B lane read (2x ds_load_b128).
template <int F>
__global__ __launch_bounds__(32) void edge_pre_wmma(
    const unsigned short* __restrict__ hbf,
    const int* __restrict__ src, const int* __restrict__ dst,
    const float* __restrict__ pre_w, const float* __restrict__ pre_b,
    float* __restrict__ asum, float* __restrict__ asq,
    unsigned* __restrict__ amin, unsigned* __restrict__ amax) {
  constexpr int CH  = 2 * F;
  constexpr int NT  = (CH + 15) / 16;
  constexpr int KT  = 3;                // K padded to 96 for the aligned layout
  constexpr int KP  = 96;
  constexpr int TPB = 16;               // edge tiles per block
  static_assert(NE % (16 * TPB) == 0, "all tiles full");
  static_assert(HID <= HBW && HBW * 2 % 16 == 0, "hbf rows 16B-aligned");

  __shared__ alignas(32) unsigned short sA[16 * KP];
  __shared__ alignas(32) unsigned short sWf[KT * NT * 512];  // fragment-packed

  const int lane = threadIdx.x;
  const int half = lane >> 4;
  const int nn   = lane & 15;

  // one-time: zero A (tail K region stays zero), stage fragment-packed weights
  for (int idx = lane; idx < 16 * KP; idx += 32) sA[idx] = 0;
  for (int idx = lane; idx < KT * NT * 512; idx += 32) {
    int elem = idx & 15;
    int ln   = (idx >> 4) & 31;
    int fr   = idx >> 9;                         // kt*NT + nt
    int kt   = fr / NT, nt = fr % NT;
    int k = kt * 32 + ((elem >> 3) << 4) + ((ln >> 4) << 3) + (elem & 7);   // A column, K_local = 16*(elem/8)+8*half+elem%8
    int c = nt * 16 + (ln & 15);
    float v = 0.0f;
    if (k < 2 * HBW && c < CH) {
      int p = k / HBW, kp = k % HBW;             // p: 0=dst half, 1=src half
      if (kp < F) v = pre_w[((c / F) * CH + (p * F + kp)) * F + (c % F)];
    }
    sWf[idx] = f2bf(v);
  }

  // per-lane column mask + bias (same for every tile)
  float biasv[NT];
  bool  cok[NT];
#pragma unroll
  for (int nt = 0; nt < NT; ++nt) {
    int c = nt * 16 + nn;
    cok[nt]   = (c < CH);
    biasv[nt] = cok[nt] ? pre_b[(c / F) * F + (c % F)] : 0.0f;
  }

  const v16s* wv = (const v16s*)sWf;

  for (int tt = 0; tt < TPB; ++tt) {
    const int e0 = (blockIdx.x * TPB + tt) * 16;
    // lane<16: dst index of edge row lane; lane>=16: src index of edge row lane-16
    const int e      = e0 + nn;
    const int mynode = (lane < 16) ? dst[e] : src[e];
    // wide 16B copies: h_bf row (80B) -> A row region [half*40, half*40+40) halfs
    {
      const uint4* grow = (const uint4*)(hbf + (size_t)mynode * HBW);
      uint4*       lrow = (uint4*)(sA + nn * KP + half * HBW);
#pragma unroll
      for (int q = 0; q < 5; ++q) lrow[q] = grow[q];
    }
    __syncthreads();

    v8f acc[NT];
    v8f z = {0.f, 0.f, 0.f, 0.f, 0.f, 0.f, 0.f, 0.f};
#pragma unroll
    for (int nt = 0; nt < NT; ++nt) acc[nt] = z;

#pragma unroll
    for (int kt = 0; kt < KT; ++kt) {
      // A fragment: lane row M = lane&15; two contiguous 16B runs
      v16s av;
#pragma unroll
      for (int j = 0; j < 8; ++j) {
        int kl   = ((j < 4) ? 0 : 16) + half * 8 + (j & 3) * 2;
        int base = nn * KP + kt * 32 + kl;
        av[2 * j]     = (short)sA[base];
        av[2 * j + 1] = (short)sA[base + 1];
      }
      v16bf af = __builtin_bit_cast(v16bf, av);
#pragma unroll
      for (int nt = 0; nt < NT; ++nt) {
        v16bf bf = __builtin_bit_cast(v16bf, wv[(kt * NT + nt) * 32 + lane]);
        acc[nt] = __builtin_amdgcn_wmma_f32_16x16x32_bf16(
            false, af, false, bf, (short)0, acc[nt], false, false);
      }
    }

    // this lane's 8 dst-row offsets via cross-lane shuffle (lanes 0..15 hold dst)
    int drow[8];
#pragma unroll
    for (int r = 0; r < 8; ++r) drow[r] = __shfl(mynode, half * 8 + r) * CH;

    // bias + scatter aggregation (D layout: vgpr r -> M = r + 8*half, col = nn)
#pragma unroll
    for (int nt = 0; nt < NT; ++nt) {
      if (cok[nt]) {
        int c = nt * 16 + nn;
#pragma unroll
        for (int r = 0; r < 8; ++r) {
          float v = acc[nt][r] + biasv[nt];
          int   o = drow[r] + c;
          atomicAdd(&asum[o], v);
          atomicAdd(&asq[o], v * v);
          unsigned u = fmap(v);
          atomicMin(&amin[o], u);
          atomicMax(&amax[o], u);
        }
      }
    }
    __syncthreads();  // protect sA before next tile's stores
  }
}

// ---------- node post-MLP + linear (+ BN stats) ----------
template <int F>
__global__ __launch_bounds__(64) void node_post(
    const float* __restrict__ h, const float* __restrict__ asum, const float* __restrict__ asq,
    const unsigned* __restrict__ amin, const unsigned* __restrict__ amax,
    const int* __restrict__ cnt, const float* __restrict__ amp, const float* __restrict__ att,
    const float* __restrict__ post_w, const float* __restrict__ post_b,
    const float* __restrict__ lin_w, const float* __restrict__ lin_b,
    float* __restrict__ hout, float* __restrict__ bn_sum, float* __restrict__ bn_sq, int do_bn) {
  constexpr int CH   = 2 * F;
  constexpr int IN13 = 13 * F;
  __shared__ float sh_in[2 * IN13];
  __shared__ float sh_post[HID];

  const int n = blockIdx.x, tid = threadIdx.x;
  float cf   = (float)cnt[n];
  float cmax = fmaxf(cf, 1.0f);
  bool  has  = cf > 0.0f;
  float a_amp = amp[n], a_att = att[n];

  for (int idx = tid; idx < 2 * IN13; idx += 64) {
    int t = idx / IN13, f = idx % IN13;
    int s = f / F, k = f % F;
    float v;
    if (s == 0) {
      v = h[n * HID + k];
    } else {
      int o = n * CH + t * F + k;
      int aggsel = (s - 1) & 3;
      float val;
      if (aggsel == 0) val = asum[o] / cmax;                       // mean
      else if (aggsel == 1) val = has ? funmap(amin[o]) : 0.0f;    // min
      else if (aggsel == 2) val = has ? funmap(amax[o]) : 0.0f;    // max
      else {                                                       // std
        float mean = asum[o] / cmax;
        float d = asq[o] / cmax - mean * mean;
        val = sqrtf(fmaxf(d, 0.0f) + 1e-5f);
      }
      int sc = (s - 1) >> 2;
      v = (sc == 0) ? val : (sc == 1 ? val * a_amp : val * a_att);
    }
    sh_in[idx] = v;
  }
  __syncthreads();
  if (tid < HID) {
    int t = tid / FOUT, g = tid % FOUT;
    const float* w  = post_w + t * IN13 * FOUT;
    const float* in = sh_in + t * IN13;
    float a = post_b[t * FOUT + g];
    for (int f = 0; f < IN13; ++f) a += in[f] * w[f * FOUT + g];
    sh_post[tid] = a;
  }
  __syncthreads();
  if (tid < HID) {
    float a = lin_b[tid];
    for (int k = 0; k < HID; ++k) a += sh_post[k] * lin_w[k * HID + tid];
    hout[n * HID + tid] = a;
    if (do_bn) {
      atomicAdd(&bn_sum[tid], a);
      atomicAdd(&bn_sq[tid], a * a);
    }
  }
}

__global__ void bn_relu(const float* __restrict__ hin, float* __restrict__ hout,
                        unsigned short* __restrict__ hbf,
                        const float* __restrict__ bsum, const float* __restrict__ bsq,
                        const float* __restrict__ g, const float* __restrict__ b) {
  int i = blockIdx.x * blockDim.x + threadIdx.x;
  if (i >= NN * HID) return;
  int n = i / HID, c = i % HID;
  float mu  = bsum[c] / (float)NN;
  float var = bsq[c] / (float)NN - mu * mu;
  float y = (hin[i] - mu) * rsqrtf(var + 1e-5f) * g[c] + b[c];
  y = fmaxf(y, 0.0f);
  hout[i] = y;
  hbf[n * HBW + c] = f2bf(y);   // pad columns stay zero (written by k_inith)
}

// ---------- final MLP + masked write-back ----------
__global__ __launch_bounds__(64) void mlp_out(
    const float* __restrict__ hin, const float* __restrict__ x,
    const int* __restrict__ nmask, const int* __restrict__ emask,
    const float* __restrict__ w0, const float* __restrict__ b0,
    const float* __restrict__ w1, const float* __restrict__ b1,
    const float* __restrict__ w2, const float* __restrict__ b2,
    const float* __restrict__ w3, const float* __restrict__ b3,
    float* __restrict__ out) {
  __shared__ float sa[HID], sb[HID];
  int n = blockIdx.x, tid = threadIdx.x;
  if (tid < HID) sa[tid] = hin[n * HID + tid];
  __syncthreads();
  if (tid < HID) { float a = b0[tid]; for (int k = 0; k < HID; ++k) a += sa[k] * w0[k * HID + tid]; sb[tid] = fmaxf(a, 0.f); }
  __syncthreads();
  if (tid < HID) { float a = b1[tid]; for (int k = 0; k < HID; ++k) a += sb[k] * w1[k * HID + tid]; sa[tid] = fmaxf(a, 0.f); }
  __syncthreads();
  if (tid < HID) { float a = b2[tid]; for (int k = 0; k < HID; ++k) a += sa[k] * w2[k * HID + tid]; sb[tid] = fmaxf(a, 0.f); }
  __syncthreads();
  if (tid < 5) {
    float a = b3[tid];
    for (int k = 0; k < HID; ++k) a += sb[k] * w3[k * 5 + tid];
    bool cond = ((nmask[n] != 0) || (emask[n] != 0)) && tid >= 1;  // cols 1..4
    out[n * 5 + tid] = cond ? a : x[n * 5 + tid];
  }
}

// ---------- host ----------
static inline int cdiv(int a, int b) { return (a + b - 1) / b; }

extern "C" void kernel_launch(void* const* d_in, const int* in_sizes, int n_in,
                              void* d_out, int out_size, void* d_ws, size_t ws_size,
                              hipStream_t stream) {
  (void)in_sizes; (void)n_in; (void)out_size; (void)ws_size;
  // inputs: dict insertion order for top level; params pytree in sorted-key order
  const float* x  = (const float*)d_in[0];
  const float* t  = (const float*)d_in[1];
  const int*   ei = (const int*)d_in[2];
  const int* src  = ei;
  const int* dst  = ei + NE;
  const int* nmask = (const int*)d_in[3];
  const int* emask = (const int*)d_in[4];
  const float* bn_b_all = (const float*)d_in[5];  // [5,34]
  const float* bn_g_all = (const float*)d_in[6];  // [5,34]
  // per-layer leaves (sorted): lin_b, lin_w, post_b, post_w, pre_b, pre_w
  auto LP = [&](int l, int slot) { return (const float*)d_in[7 + l * 6 + slot]; };
  const float* mlp_b[4]; const float* mlp_w[4];
  for (int i = 0; i < 4; ++i) { mlp_b[i] = (const float*)d_in[43 + 2 * i]; mlp_w[i] = (const float*)d_in[44 + 2 * i]; }

  // workspace carve (~44 MB)
  size_t off = 0;
  auto carve = [&](size_t bytes) { void* p = (char*)d_ws + off; off += (bytes + 255) & ~(size_t)255; return p; };
  int*            cnt    = (int*)carve(NN * 4);
  float*          lsum   = (float*)carve(4);
  float*          amp    = (float*)carve(NN * 4);
  float*          att    = (float*)carve(NN * 4);
  float*          h_cur  = (float*)carve((size_t)NN * HID * 4);
  float*          h_nxt  = (float*)carve((size_t)NN * HID * 4);
  unsigned short* h_bf   = (unsigned short*)carve((size_t)NN * HBW * 2);
  float*          asum   = (float*)carve((size_t)NN * 68 * 4);
  float*          asq    = (float*)carve((size_t)NN * 68 * 4);
  unsigned*       amin   = (unsigned*)carve((size_t)NN * 68 * 4);
  unsigned*       amax   = (unsigned*)carve((size_t)NN * 68 * 4);
  float*          bn_sum = (float*)carve(HID * 4);
  float*          bn_sq  = (float*)carve(HID * 4);

  hipMemsetAsync(cnt, 0, NN * 4, stream);
  hipMemsetAsync(lsum, 0, 4, stream);
  k_count<<<cdiv(NE, 256), 256, 0, stream>>>(dst, cnt);
  k_logsum<<<cdiv(NN, 256), 256, 0, stream>>>(cnt, lsum);
  k_degscale<<<cdiv(NN, 256), 256, 0, stream>>>(cnt, lsum, amp, att);
  k_inith<<<cdiv(NN * HBW, 256), 256, 0, stream>>>(x, t, h_cur, h_bf);

  for (int l = 0; l < NDEP; ++l) {
    const int CH = (l == 0) ? 12 : 68;
    hipMemsetAsync(asum, 0, (size_t)NN * CH * 4, stream);
    hipMemsetAsync(asq, 0, (size_t)NN * CH * 4, stream);
    k_init_minmax<<<cdiv(NN * CH, 256), 256, 0, stream>>>(amin, amax, NN * CH);

    const int edge_blocks = NE / 256;  // 16 tiles x 16 edges per block, exact
    if (l == 0)
      edge_pre_wmma<6><<<edge_blocks, 32, 0, stream>>>(h_bf, src, dst, LP(l, 5), LP(l, 4), asum, asq, amin, amax);
    else
      edge_pre_wmma<34><<<edge_blocks, 32, 0, stream>>>(h_bf, src, dst, LP(l, 5), LP(l, 4), asum, asq, amin, amax);

    int do_bn = (l < NDEP - 1) ? 1 : 0;
    if (do_bn) {
      hipMemsetAsync(bn_sum, 0, HID * 4, stream);
      hipMemsetAsync(bn_sq, 0, HID * 4, stream);
    }
    if (l == 0)
      node_post<6><<<NN, 64, 0, stream>>>(h_cur, asum, asq, amin, amax, cnt, amp, att,
                                          LP(l, 3), LP(l, 2), LP(l, 1), LP(l, 0),
                                          h_nxt, bn_sum, bn_sq, do_bn);
    else
      node_post<34><<<NN, 64, 0, stream>>>(h_cur, asum, asq, amin, amax, cnt, amp, att,
                                           LP(l, 3), LP(l, 2), LP(l, 1), LP(l, 0),
                                           h_nxt, bn_sum, bn_sq, do_bn);
    if (do_bn)
      bn_relu<<<cdiv(NN * HID, 256), 256, 0, stream>>>(h_nxt, h_cur, h_bf, bn_sum, bn_sq,
                                                       bn_g_all + l * HID, bn_b_all + l * HID);
  }

  mlp_out<<<NN, 64, 0, stream>>>(h_nxt, x, nmask, emask,
                                 mlp_w[0], mlp_b[0], mlp_w[1], mlp_b[1],
                                 mlp_w[2], mlp_b[2], mlp_w[3], mlp_b[3],
                                 (float*)d_out);
}
